// MEGNet_37855841747615
// MI455X (gfx1250) — compile-verified, matrix operand order checked by reading, not run on hardware
//
#include <hip/hip_runtime.h>
#include <hip/hip_bf16.h>
#include <math.h>

// ---------------------------------------------------------------------------
// MEGNet on MI455X (gfx1250): all GEMMs on v_wmma_f32_16x16x32_bf16, wave32,
// one wave per 16-row tile (EXEC all ones for WMMA).
// ---------------------------------------------------------------------------

typedef __bf16 bf16_t;
typedef __attribute__((ext_vector_type(16))) __bf16 bf16x16;
typedef __attribute__((ext_vector_type(8)))  float  f32x8;

#define WMMA_BF16(a, b, c) \
  __builtin_amdgcn_wmma_f32_16x16x32_bf16(false, (a), false, (b), (short)0, (c), false, false)

__device__ __forceinline__ float softplus_f(float v) {
  return (v > 20.0f) ? v : log1pf(__expf(v));
}

// A-fragment from two 8-float global/generic chunks (row-major source rows).
// Layout: lanes 0-15 hold K=[k0..k0+7] in f[0..7], K=[k0+16..k0+23] in f[8..15];
// lanes 16-31 use k0+8 (handled by caller via `hi`).
__device__ __forceinline__ bf16x16 frag_f32(const float* p0, const float* p1) {
  bf16x16 f;
#pragma unroll
  for (int i = 0; i < 8; ++i) f[i] = (bf16_t)p0[i];
#pragma unroll
  for (int i = 0; i < 8; ++i) f[8 + i] = (bf16_t)p1[i];
  return f;
}

// B-fragment from bf16 W^T (row-major [64, K]); lane holds column n of W.
__device__ __forceinline__ bf16x16 frag_w(const bf16_t* wt, int K, int ntile, int ktile, int lane) {
  const bf16_t* r = wt + (size_t)(ntile * 16 + (lane & 15)) * K + ktile * 32 + ((lane & 16) ? 8 : 0);
  bf16x16 f;
#pragma unroll
  for (int i = 0; i < 8; ++i) f[i] = r[i];
#pragma unroll
  for (int i = 0; i < 8; ++i) f[8 + i] = r[16 + i];
  return f;
}

// A-fragment from an LDS bf16 row (16B-aligned chunks).
__device__ __forceinline__ bf16x16 frag_lds(const bf16_t* base) {
  bf16x16 f;
#pragma unroll
  for (int i = 0; i < 8; ++i) f[i] = base[i];
#pragma unroll
  for (int i = 0; i < 8; ++i) f[8 + i] = base[16 + i];
  return f;
}

// ---------------------------------------------------------------------------
// Weight prep: f32 [K,64] row-major -> bf16 [64,K] (W transpose), per launch.
// ---------------------------------------------------------------------------
__global__ void k_transpose(const float* __restrict__ in, bf16_t* __restrict__ out, int K) {
  int idx = blockIdx.x * blockDim.x + threadIdx.x;
  if (idx >= K * 64) return;
  int n = idx / K;
  int k = idx - n * K;
  out[idx] = (bf16_t)in[k * 64 + n];
}

// ---------------------------------------------------------------------------
// Encoders
// ---------------------------------------------------------------------------
__global__ __launch_bounds__(32) void k_enc_node(const float* __restrict__ xin,   // [N,32]
                                                 const bf16_t* __restrict__ WT,   // [64,32]
                                                 const float* __restrict__ b,
                                                 float* __restrict__ xout) {      // [N,64]
  __shared__ __align__(16) float s_o[16 * 64];
  const int lane = threadIdx.x;
  const int base = blockIdx.x * 16;
  const int m = lane & 15;
  const int hi = (lane & 16) ? 8 : 0;
  const float* row = xin + (size_t)(base + m) * 32;
  bf16x16 a = frag_f32(row + hi, row + 16 + hi);
  f32x8 acc[4] = {};
#pragma unroll
  for (int nt = 0; nt < 4; ++nt) acc[nt] = WMMA_BF16(a, frag_w(WT, 32, nt, 0, lane), acc[nt]);
#pragma unroll
  for (int nt = 0; nt < 4; ++nt) {
    int n = (lane & 15) + nt * 16;
    float bb = b[n];
#pragma unroll
    for (int v = 0; v < 8; ++v) s_o[(v + hi) * 64 + n] = acc[nt][v] + bb;
  }
  __syncthreads();
  int me = lane >> 1, half = (lane & 1) * 32;
  float* op = xout + (size_t)(base + me) * 64 + half;
  const float* sp = s_o + me * 64 + half;
#pragma unroll
  for (int j = 0; j < 32; j += 4) *(float4*)(op + j) = *(const float4*)(sp + j);
}

__global__ __launch_bounds__(32) void k_enc_edge(const float* __restrict__ attr,  // [E,16]
                                                 const bf16_t* __restrict__ WT,   // [64,16]
                                                 const float* __restrict__ b,
                                                 float* __restrict__ eout) {      // [E,64]
  __shared__ __align__(16) float s_o[16 * 64];
  const int lane = threadIdx.x;
  const int base = blockIdx.x * 16;
  const int m = lane & 15;
  const int hi = (lane & 16) ? 8 : 0;
  const float* row = attr + (size_t)(base + m) * 16;
  bf16x16 a;
#pragma unroll
  for (int i = 0; i < 8; ++i) a[i] = (bf16_t)row[hi + i];
#pragma unroll
  for (int i = 0; i < 8; ++i) a[8 + i] = (bf16_t)0.0f;   // K padded 16->32 with zeros
  f32x8 acc[4] = {};
#pragma unroll
  for (int nt = 0; nt < 4; ++nt) {
    const bf16_t* r = WT + (size_t)(nt * 16 + (lane & 15)) * 16 + hi;
    bf16x16 bw;
#pragma unroll
    for (int i = 0; i < 8; ++i) bw[i] = r[i];
#pragma unroll
    for (int i = 0; i < 8; ++i) bw[8 + i] = (bf16_t)0.0f;
    acc[nt] = WMMA_BF16(a, bw, acc[nt]);
  }
#pragma unroll
  for (int nt = 0; nt < 4; ++nt) {
    int n = (lane & 15) + nt * 16;
    float bb = b[n];
#pragma unroll
    for (int v = 0; v < 8; ++v) s_o[(v + hi) * 64 + n] = acc[nt][v] + bb;
  }
  __syncthreads();
  int me = lane >> 1, half = (lane & 1) * 32;
  float* op = eout + (size_t)(base + me) * 64 + half;
  const float* sp = s_o + me * 64 + half;
#pragma unroll
  for (int j = 0; j < 32; j += 4) *(float4*)(op + j) = *(const float4*)(sp + j);
}

__global__ void k_enc_glob(const float* __restrict__ gs,     // [16]
                           const float* __restrict__ W,      // [16,64]
                           const float* __restrict__ b,
                           float* __restrict__ g) {          // [64]
  int t = threadIdx.x;  // 64 threads
  float a = b[t];
#pragma unroll
  for (int k = 0; k < 16; ++k) a += gs[k] * W[k * 64 + t];
  g[t] = a;
}

// ---------------------------------------------------------------------------
// Edge MLP block: e_new = MLP([e, x[row], x[col]]); e += e_new;
// msg[row] += e_new (segment sum); e_sum += colsum(e_new).
// ---------------------------------------------------------------------------
__global__ __launch_bounds__(32) void k_edge_block(
    float* e, const float* __restrict__ x, const int* __restrict__ ei,
    const bf16_t* __restrict__ W1T, const float* __restrict__ b1,
    const bf16_t* __restrict__ W2T, const float* __restrict__ b2,
    float* __restrict__ msg, float* __restrict__ e_sum, int E) {
  __shared__ __align__(16) bf16_t s_h[16 * 64];
  __shared__ __align__(16) float  s_en[16 * 64];
  const int lane = threadIdx.x;
  const int base = blockIdx.x * 16;
  const int m = lane & 15;
  const int hi = (lane & 16) ? 8 : 0;
  const int eidx = base + m;
  const int r = ei[eidx];
  const int c = ei[E + eidx];
  const float* erow = e + (size_t)eidx * 64;
  const float* xr = x + (size_t)r * 64;
  const float* xc = x + (size_t)c * 64;

  f32x8 acc[4] = {};
#pragma unroll
  for (int kt = 0; kt < 6; ++kt) {               // K = 192 = concat(e, x[row], x[col])
    int k0 = kt * 32 + hi;
    int k1 = k0 + 16;
    const float* p0 = (k0 < 64) ? (erow + k0) : (k0 < 128) ? (xr + (k0 - 64)) : (xc + (k0 - 128));
    const float* p1 = (k1 < 64) ? (erow + k1) : (k1 < 128) ? (xr + (k1 - 64)) : (xc + (k1 - 128));
    bf16x16 a = frag_f32(p0, p1);
#pragma unroll
    for (int nt = 0; nt < 4; ++nt) acc[nt] = WMMA_BF16(a, frag_w(W1T, 192, nt, kt, lane), acc[nt]);
  }
#pragma unroll
  for (int nt = 0; nt < 4; ++nt) {               // bias + softplus -> bf16 LDS (transpose layout)
    int n = (lane & 15) + nt * 16;
    float bb = b1[n];
#pragma unroll
    for (int v = 0; v < 8; ++v) s_h[(v + hi) * 64 + n] = (bf16_t)softplus_f(acc[nt][v] + bb);
  }
  __syncthreads();

  f32x8 acc2[4] = {};
#pragma unroll
  for (int kt = 0; kt < 2; ++kt) {               // K = 64
    bf16x16 a = frag_lds(s_h + m * 64 + kt * 32 + hi);
#pragma unroll
    for (int nt = 0; nt < 4; ++nt) acc2[nt] = WMMA_BF16(a, frag_w(W2T, 64, nt, kt, lane), acc2[nt]);
  }
#pragma unroll
  for (int nt = 0; nt < 4; ++nt) {
    int n = (lane & 15) + nt * 16;
    float bb = b2[n];
#pragma unroll
    for (int v = 0; v < 8; ++v) s_en[(v + hi) * 64 + n] = acc2[nt][v] + bb;
  }
  __syncthreads();

  {  // residual e += e_new (coalesced) + segment-sum scatter into msg
    int me = lane >> 1;
    int half = (lane & 1) * 32;
    float* ep = e + (size_t)(base + me) * 64 + half;
    const float* sp = s_en + me * 64 + half;
#pragma unroll
    for (int j = 0; j < 32; j += 4) {
      float4 o = *(const float4*)(ep + j);
      float4 d = *(const float4*)(sp + j);
      o.x += d.x; o.y += d.y; o.z += d.z; o.w += d.w;
      *(float4*)(ep + j) = o;
    }
    int rr = ei[base + me];
    float* mp = msg + (size_t)rr * 64 + half;
#pragma unroll
    for (int j = 0; j < 32; ++j) atomicAdd(mp + j, sp[j]);
  }
  {  // column sums of e_new for the global mean
    float s0 = 0.f, s1 = 0.f;
#pragma unroll
    for (int mm = 0; mm < 16; ++mm) { s0 += s_en[mm * 64 + lane]; s1 += s_en[mm * 64 + lane + 32]; }
    atomicAdd(e_sum + lane, s0);
    atomicAdd(e_sum + lane + 32, s1);
  }
}

// ---------------------------------------------------------------------------
// Node MLP block: x_new = MLP([x, msg]); x += x_new; x_sum += colsum(x_new).
// ---------------------------------------------------------------------------
__global__ __launch_bounds__(32) void k_node_block(
    float* x, const float* __restrict__ msg,
    const bf16_t* __restrict__ W1T, const float* __restrict__ b1,
    const bf16_t* __restrict__ W2T, const float* __restrict__ b2,
    float* __restrict__ x_sum) {
  __shared__ __align__(16) bf16_t s_h[16 * 64];
  __shared__ __align__(16) float  s_xn[16 * 64];
  const int lane = threadIdx.x;
  const int base = blockIdx.x * 16;
  const int m = lane & 15;
  const int hi = (lane & 16) ? 8 : 0;
  const float* xrow = x + (size_t)(base + m) * 64;
  const float* mrow = msg + (size_t)(base + m) * 64;

  f32x8 acc[4] = {};
#pragma unroll
  for (int kt = 0; kt < 4; ++kt) {               // K = 128 = concat(x, msg)
    int k0 = kt * 32 + hi;
    int k1 = k0 + 16;
    const float* p0 = (k0 < 64) ? (xrow + k0) : (mrow + (k0 - 64));
    const float* p1 = (k1 < 64) ? (xrow + k1) : (mrow + (k1 - 64));
    bf16x16 a = frag_f32(p0, p1);
#pragma unroll
    for (int nt = 0; nt < 4; ++nt) acc[nt] = WMMA_BF16(a, frag_w(W1T, 128, nt, kt, lane), acc[nt]);
  }
#pragma unroll
  for (int nt = 0; nt < 4; ++nt) {
    int n = (lane & 15) + nt * 16;
    float bb = b1[n];
#pragma unroll
    for (int v = 0; v < 8; ++v) s_h[(v + hi) * 64 + n] = (bf16_t)softplus_f(acc[nt][v] + bb);
  }
  __syncthreads();

  f32x8 acc2[4] = {};
#pragma unroll
  for (int kt = 0; kt < 2; ++kt) {
    bf16x16 a = frag_lds(s_h + m * 64 + kt * 32 + hi);
#pragma unroll
    for (int nt = 0; nt < 4; ++nt) acc2[nt] = WMMA_BF16(a, frag_w(W2T, 64, nt, kt, lane), acc2[nt]);
  }
#pragma unroll
  for (int nt = 0; nt < 4; ++nt) {
    int n = (lane & 15) + nt * 16;
    float bb = b2[n];
#pragma unroll
    for (int v = 0; v < 8; ++v) s_xn[(v + hi) * 64 + n] = acc2[nt][v] + bb;
  }
  __syncthreads();

  {  // residual x += x_new (coalesced)
    int me = lane >> 1;
    int half = (lane & 1) * 32;
    float* xp = x + (size_t)(base + me) * 64 + half;
    const float* sp = s_xn + me * 64 + half;
#pragma unroll
    for (int j = 0; j < 32; j += 4) {
      float4 o = *(const float4*)(xp + j);
      float4 d = *(const float4*)(sp + j);
      o.x += d.x; o.y += d.y; o.z += d.z; o.w += d.w;
      *(float4*)(xp + j) = o;
    }
  }
  {  // column sums of x_new for global mean
    float s0 = 0.f, s1 = 0.f;
#pragma unroll
    for (int mm = 0; mm < 16; ++mm) { s0 += s_xn[mm * 64 + lane]; s1 += s_xn[mm * 64 + lane + 32]; }
    atomicAdd(x_sum + lane, s0);
    atomicAdd(x_sum + lane + 32, s1);
  }
}

// ---------------------------------------------------------------------------
// Global MLP block: g += MLP([x_new.mean, e_new.mean, g]); scalar (tiny).
// ---------------------------------------------------------------------------
__global__ void k_glob_block(const float* __restrict__ x_sum, const float* __restrict__ e_sum,
                             float* __restrict__ g,
                             const float* __restrict__ W1, const float* __restrict__ b1,
                             const float* __restrict__ W2, const float* __restrict__ b2,
                             float invN, float invE) {
  __shared__ float gin[192];
  __shared__ float h[64];
  int t = threadIdx.x;  // 64 threads
  gin[t]       = x_sum[t] * invN;
  gin[64 + t]  = e_sum[t] * invE;
  gin[128 + t] = g[t];
  __syncthreads();
  float a = b1[t];
  for (int k = 0; k < 192; ++k) a += gin[k] * W1[k * 64 + t];
  h[t] = softplus_f(a);
  __syncthreads();
  float o = b2[t];
  for (int k = 0; k < 64; ++k) o += h[k] * W2[k * 64 + t];
  g[t] += o;
}

// ---------------------------------------------------------------------------
// Final column-sum of x and readout head.
// ---------------------------------------------------------------------------
__global__ void k_colsum(const float* __restrict__ x, float* __restrict__ out, int N) {
  int gid = blockIdx.x * blockDim.x + threadIdx.x;
  int col = gid & 63;
  int rg = gid >> 6;
  int step = (gridDim.x * blockDim.x) >> 6;
  float s = 0.f;
  for (int r = rg; r < N; r += step) s += x[(size_t)r * 64 + col];
  atomicAdd(out + col, s);
}

__global__ void k_readout(const float* __restrict__ xfin_sum, const float* __restrict__ g,
                          const float* __restrict__ W0, const float* __restrict__ b0,
                          const float* __restrict__ g0, const float* __restrict__ be0,
                          const float* __restrict__ W1, const float* __restrict__ b1,
                          const float* __restrict__ g1, const float* __restrict__ be1,
                          const float* __restrict__ fW, const float* __restrict__ fb,
                          float* __restrict__ out, float invN) {
  __shared__ float h0[128];
  __shared__ float h1[64];
  __shared__ float h2[64];
  const float inv = rsqrtf(1.0f + 1.0e-3f);  // BN fresh stats, keras eps
  int t = threadIdx.x;  // 64 threads
  h0[t] = xfin_sum[t] * invN;
  h0[64 + t] = g[t];
  __syncthreads();
  float a = b0[t];
  for (int k = 0; k < 128; ++k) a += h0[k] * W0[k * 64 + t];
  h1[t] = g0[t] * (softplus_f(a) * inv) + be0[t];
  __syncthreads();
  a = b1[t];
  for (int k = 0; k < 64; ++k) a += h1[k] * W1[k * 64 + t];
  h2[t] = g1[t] * (softplus_f(a) * inv) + be1[t];
  __syncthreads();
  if (t < 3) {
    float o = fb[t];
    for (int k = 0; k < 64; ++k) o += h2[k] * fW[k * 3 + t];
    out[t] = o;
  }
}

// ---------------------------------------------------------------------------
extern "C" void kernel_launch(void* const* d_in, const int* in_sizes, int n_in,
                              void* d_out, int out_size, void* d_ws, size_t ws_size,
                              hipStream_t stream) {
  (void)n_in; (void)out_size; (void)ws_size;
  const float* x_in   = (const float*)d_in[0];
  const int*   ei     = (const int*)d_in[1];
  const float* e_attr = (const float*)d_in[2];
  const float* gs     = (const float*)d_in[3];
  const float* node_W = (const float*)d_in[4];
  const float* node_b = (const float*)d_in[5];
  const float* edge_W = (const float*)d_in[6];
  const float* edge_b = (const float*)d_in[7];
  const float* glob_W = (const float*)d_in[8];
  const float* glob_b = (const float*)d_in[9];
  const float* be_W1  = (const float*)d_in[10];
  const float* be_b1  = (const float*)d_in[11];
  const float* be_W2  = (const float*)d_in[12];
  const float* be_b2  = (const float*)d_in[13];
  const float* bn_W1  = (const float*)d_in[14];
  const float* bn_b1  = (const float*)d_in[15];
  const float* bn_W2  = (const float*)d_in[16];
  const float* bn_b2  = (const float*)d_in[17];
  const float* bg_W1  = (const float*)d_in[18];
  const float* bg_b1  = (const float*)d_in[19];
  const float* bg_W2  = (const float*)d_in[20];
  const float* bg_b2  = (const float*)d_in[21];
  const float* out_W0 = (const float*)d_in[22];
  const float* out_b0 = (const float*)d_in[23];
  const float* out_g0 = (const float*)d_in[24];
  const float* out_be0= (const float*)d_in[25];
  const float* out_W1 = (const float*)d_in[26];
  const float* out_b1 = (const float*)d_in[27];
  const float* out_g1 = (const float*)d_in[28];
  const float* out_be1= (const float*)d_in[29];
  const float* fin_W  = (const float*)d_in[30];
  const float* fin_b  = (const float*)d_in[31];

  const int N = in_sizes[0] / 32;   // 50000
  const int E = in_sizes[2] / 16;   // 400000

  // Workspace carve-out (256B aligned regions)
  char* ws = (char*)d_ws;
  size_t off = 0;
  auto carve = [&](size_t bytes) -> void* {
    off = (off + 255) & ~(size_t)255;
    void* p = ws + off;
    off += bytes;
    return p;
  };
  float* xbuf   = (float*)carve((size_t)N * 64 * 4);
  float* ebuf   = (float*)carve((size_t)E * 64 * 4);
  float* msg    = (float*)carve((size_t)N * 64 * 4);
  float* gbuf   = (float*)carve(64 * 4);
  float* e_sum  = (float*)carve(64 * 4);
  float* x_sum  = (float*)carve(64 * 4);
  float* xfin   = (float*)carve(64 * 4);
  bf16_t* node_WT = (bf16_t*)carve(64 * 32 * 2);
  bf16_t* edge_WT = (bf16_t*)carve(64 * 16 * 2);
  bf16_t* beT1    = (bf16_t*)carve((size_t)3 * 64 * 192 * 2);
  bf16_t* beT2    = (bf16_t*)carve((size_t)3 * 64 * 64 * 2);
  bf16_t* bnT1    = (bf16_t*)carve((size_t)3 * 64 * 128 * 2);
  bf16_t* bnT2    = (bf16_t*)carve((size_t)3 * 64 * 64 * 2);

  auto T = [&](const float* src, bf16_t* dst, int K) {
    int total = K * 64;
    k_transpose<<<(total + 255) / 256, 256, 0, stream>>>(src, dst, K);
  };
  T(node_W, node_WT, 32);
  T(edge_W, edge_WT, 16);
  for (int i = 0; i < 3; ++i) {
    T(be_W1 + (size_t)i * 192 * 64, beT1 + (size_t)i * 64 * 192, 192);
    T(be_W2 + (size_t)i * 64 * 64,  beT2 + (size_t)i * 64 * 64,  64);
    T(bn_W1 + (size_t)i * 128 * 64, bnT1 + (size_t)i * 64 * 128, 128);
    T(bn_W2 + (size_t)i * 64 * 64,  bnT2 + (size_t)i * 64 * 64,  64);
  }

  // Encoders
  k_enc_node<<<N / 16, 32, 0, stream>>>(x_in, node_WT, node_b, xbuf);
  k_enc_edge<<<E / 16, 32, 0, stream>>>(e_attr, edge_WT, edge_b, ebuf);
  k_enc_glob<<<1, 64, 0, stream>>>(gs, glob_W, glob_b, gbuf);

  const float invN = 1.0f / (float)N;
  const float invE = 1.0f / (float)E;

  for (int i = 0; i < 3; ++i) {
    hipMemsetAsync(msg,   0, (size_t)N * 64 * 4, stream);
    hipMemsetAsync(e_sum, 0, 64 * 4, stream);
    hipMemsetAsync(x_sum, 0, 64 * 4, stream);
    k_edge_block<<<E / 16, 32, 0, stream>>>(
        ebuf, xbuf, ei,
        beT1 + (size_t)i * 64 * 192, be_b1 + i * 64,
        beT2 + (size_t)i * 64 * 64,  be_b2 + i * 64,
        msg, e_sum, E);
    k_node_block<<<N / 16, 32, 0, stream>>>(
        xbuf, msg,
        bnT1 + (size_t)i * 64 * 128, bn_b1 + i * 64,
        bnT2 + (size_t)i * 64 * 64,  bn_b2 + i * 64,
        x_sum);
    k_glob_block<<<1, 64, 0, stream>>>(
        x_sum, e_sum, gbuf,
        bg_W1 + (size_t)i * 192 * 64, bg_b1 + i * 64,
        bg_W2 + (size_t)i * 64 * 64,  bg_b2 + i * 64,
        invN, invE);
  }

  hipMemsetAsync(xfin, 0, 64 * 4, stream);
  k_colsum<<<64, 256, 0, stream>>>(xbuf, xfin, N);
  k_readout<<<1, 64, 0, stream>>>(xfin, gbuf,
                                  out_W0, out_b0, out_g0, out_be0,
                                  out_W1, out_b1, out_g1, out_be1,
                                  fin_W, fin_b, (float*)d_out, invN);
}
